// GraphAttention_23527830847950
// MI455X (gfx1250) — matrix operand order repaired and checked by reading, not compiled, not run on hardware
//
#include <hip/hip_runtime.h>
#include <hip/hip_bf16.h>

#define N_NODES 100000
#define N_EDGES 1600000
#define IN_DIM  256
#define HEADS   4
#define OUT_DIM 64
#define HD      256          // HEADS*OUT_DIM, == flattened channel count
#define ALPHA   0.2f

typedef __attribute__((ext_vector_type(16))) __bf16 v16bf;
typedef __attribute__((ext_vector_type(8)))  __bf16 v8bf;
typedef __attribute__((ext_vector_type(8)))  float  v8f;

// ---------------------------------------------------------------------------
// Guaranteed fire-and-forget f32 atomic add (no return -> STOREcnt only).
// Inline asm pins the CDNA5 instruction; implicit S_WAIT_IDLE at S_ENDPGM
// guarantees completion before the kernel retires.
// ---------------------------------------------------------------------------
__device__ __forceinline__ void atomic_add_f32_nr(float* p, float v) {
#if defined(__gfx1250__)
  asm volatile("global_atomic_add_f32 %0, %1, off scope:SCOPE_DEV"
               :
               : "v"((unsigned long long)(uintptr_t)p), "v"(v)
               : "memory");
#else
  atomicAdd(p, v);
#endif
}

// ---------------------------------------------------------------------------
// fp32 -> bf16 conversion (grid-stride)
// ---------------------------------------------------------------------------
__global__ void cvt_to_bf16(const float* __restrict__ in, __bf16* __restrict__ out, int n) {
  int i = blockIdx.x * blockDim.x + threadIdx.x;
  int stride = gridDim.x * blockDim.x;
  for (; i < n; i += stride) out[i] = (__bf16)in[i];
}

// ---------------------------------------------------------------------------
// zero fill (grid-stride)
// ---------------------------------------------------------------------------
__global__ void zero_f32(float* __restrict__ p, int n) {
  int i = blockIdx.x * blockDim.x + threadIdx.x;
  int stride = gridDim.x * blockDim.x;
  for (; i < n; i += stride) p[i] = 0.0f;
}

// ---------------------------------------------------------------------------
// ft = x @ W^T via v_wmma_f32_16x16x32_bf16.
// One wave computes a 16x64 strip: 4 accumulators share one A fragment
// (4x less Xb read traffic than tile-per-wave). K-loop = 256/32 = 8 steps,
// 4 WMMAs each -> 32 WMMAs per wave.
//   A tile (16x32, bf16): lanes 0-15 (grp=0) hold K 0..15,
//     lanes 16-31 (grp=1) hold K 16..31 of row M = lane&15.
//   B tile (32x16, bf16): B[k,n] = W[col+n, kk+k]; lane n (grp=0) holds
//     K 0..15 of column n (contiguous chunk of W row), lane n+16 holds K 16..31.
// ---------------------------------------------------------------------------
__global__ void __launch_bounds__(256)
gat_gemm_wmma(const __bf16* __restrict__ Xb, const __bf16* __restrict__ Wb,
              float* __restrict__ ft) {
  const int wave = threadIdx.x >> 5;
  const int lane = threadIdx.x & 31;
  const int wid  = blockIdx.x * 8 + wave;   // 6250 * 4 = 25000 waves exactly
  const int rowTile = wid >> 2;             // 0..6249
  const int cg      = wid & 3;              // 64-column group
  const int row0 = rowTile << 4;
  const int col0 = cg << 6;
  const int m   = lane & 15;
  const int grp = lane >> 4;

  const __bf16* arow = Xb + (size_t)(row0 + m) * IN_DIM;
  const __bf16* b0 = Wb + (size_t)(col0 +  0 + m) * IN_DIM;
  const __bf16* b1 = Wb + (size_t)(col0 + 16 + m) * IN_DIM;
  const __bf16* b2 = Wb + (size_t)(col0 + 32 + m) * IN_DIM;
  const __bf16* b3 = Wb + (size_t)(col0 + 48 + m) * IN_DIM;

  v8f acc0 = {}, acc1 = {}, acc2 = {}, acc3 = {};
#pragma unroll
  for (int kk = 0; kk < IN_DIM; kk += 32) {
    v16bf a = *(const v16bf*)(arow + kk + grp * 16);       // K = grp*16 + 0..15
    v16bf B0 = *(const v16bf*)(b0 + kk + grp * 16);        // K = grp*16 + 0..15
    v16bf B1 = *(const v16bf*)(b1 + kk + grp * 16);
    v16bf B2 = *(const v16bf*)(b2 + kk + grp * 16);
    v16bf B3 = *(const v16bf*)(b3 + kk + grp * 16);
    acc0 = __builtin_amdgcn_wmma_f32_16x16x32_bf16(false, a, false, B0, (short)0, acc0, false, false);
    acc1 = __builtin_amdgcn_wmma_f32_16x16x32_bf16(false, a, false, B1, (short)0, acc1, false, false);
    acc2 = __builtin_amdgcn_wmma_f32_16x16x32_bf16(false, a, false, B2, (short)0, acc2, false, false);
    acc3 = __builtin_amdgcn_wmma_f32_16x16x32_bf16(false, a, false, B3, (short)0, acc3, false, false);
  }

  // C/D layout: VGPR r -> M = r + grp*8; column = lane&15
  float* crow = ft + (size_t)(row0 + grp * 8) * HD + col0 + m;
#pragma unroll
  for (int r = 0; r < 8; ++r) {
    crow[(size_t)r * HD +  0] = acc0[r];
    crow[(size_t)r * HD + 16] = acc1[r];
    crow[(size_t)r * HD + 32] = acc2[r];
    crow[(size_t)r * HD + 48] = acc3[r];
  }
}

// ---------------------------------------------------------------------------
// Per-node, per-head attention scalars: a1[n,h] = ft[n,h,:].attn_l[h,:], a2 same
// ---------------------------------------------------------------------------
__global__ void attn_scores(const float* __restrict__ ft,
                            const float* __restrict__ attn_l,
                            const float* __restrict__ attn_r,
                            float* __restrict__ a1, float* __restrict__ a2) {
  int idx = blockIdx.x * blockDim.x + threadIdx.x;   // n*HEADS + h
  if (idx >= N_NODES * HEADS) return;
  const int n = idx >> 2;
  const int h = idx & 3;
  const float* f  = ft + (size_t)n * HD + h * OUT_DIM;
  const float* al = attn_l + h * OUT_DIM;
  const float* ar = attn_r + h * OUT_DIM;
  float s1 = 0.0f, s2 = 0.0f;
#pragma unroll 8
  for (int d = 0; d < OUT_DIM; ++d) {
    float v = f[d];
    s1 += v * al[d];
    s2 += v * ar[d];
  }
  a1[idx] = s1;
  a2[idx] = s2;
}

// ---------------------------------------------------------------------------
// Edge phase: one wave per edge. Lane l handles channels [l*8, l*8+8) of the
// flattened [H*D]=256 vector, so head h = l>>3. Gather ft[src] (2x b128),
// scale by a, scatter via no-return global_atomic_add_f32 into num[dst]
// (both ft and num fit in the 192MB L2 -> this is an L2-bandwidth problem).
// a = clip(exp(leakyrelu(a1[src]+a2[dst])), -10, 10) == min(exp(.), 10).
// ---------------------------------------------------------------------------
__global__ void __launch_bounds__(256)
edge_scatter(const float* __restrict__ ft,
             const float* __restrict__ a1, const float* __restrict__ a2,
             const int* __restrict__ src, const int* __restrict__ dst,
             float* __restrict__ num, float* __restrict__ den) {
  const int wave = threadIdx.x >> 5;
  const int lane = threadIdx.x & 31;
  const int e = blockIdx.x * 8 + wave;        // N_EDGES/8 blocks -> exact cover
  const int s = src[e];
  const int d = dst[e];
  const int h = lane >> 3;

  float sc = a1[s * HEADS + h] + a2[d * HEADS + h];
  sc = sc > 0.0f ? sc : ALPHA * sc;
  float a = fminf(__expf(sc), 10.0f);

  const float4* frow = (const float4*)(ft + (size_t)s * HD) + lane * 2;
  float4 f0 = frow[0];
  float4 f1 = frow[1];

  float* nrow = num + (size_t)d * HD + lane * 8;
  atomic_add_f32_nr(nrow + 0, a * f0.x);
  atomic_add_f32_nr(nrow + 1, a * f0.y);
  atomic_add_f32_nr(nrow + 2, a * f0.z);
  atomic_add_f32_nr(nrow + 3, a * f0.w);
  atomic_add_f32_nr(nrow + 4, a * f1.x);
  atomic_add_f32_nr(nrow + 5, a * f1.y);
  atomic_add_f32_nr(nrow + 6, a * f1.z);
  atomic_add_f32_nr(nrow + 7, a * f1.w);

  if ((lane & 7) == 0) atomic_add_f32_nr(den + (size_t)d * HEADS + h, a);
}

// ---------------------------------------------------------------------------
// out[n,h,d] = num[n,h,d] / den[n,h]   (in place on d_out)
// ---------------------------------------------------------------------------
__global__ void normalize_out(float* __restrict__ out, const float* __restrict__ den) {
  int i = blockIdx.x * blockDim.x + threadIdx.x;   // over N*HD
  if (i >= N_NODES * HD) return;
  const int n = i >> 8;        // /256
  const int c = i & 255;
  const int h = c >> 6;        // /64
  out[i] = out[i] / den[n * HEADS + h];
}

// ---------------------------------------------------------------------------
// Workspace layout (all region sizes multiples of 256B, base hipMalloc-aligned):
//   Xb  : N*256 bf16           51,200,000 B
//   Wb  : 256*256 bf16            131,072 B
//   ft  : N*256 f32           102,400,000 B
//   a1  : N*4 f32               1,600,000 B
//   a2  : N*4 f32               1,600,000 B
//   den : N*4 f32               1,600,000 B      total ~158.5 MB
// ---------------------------------------------------------------------------
extern "C" void kernel_launch(void* const* d_in, const int* in_sizes, int n_in,
                              void* d_out, int out_size, void* d_ws, size_t ws_size,
                              hipStream_t stream) {
  const float* x      = (const float*)d_in[0];
  const float* W      = (const float*)d_in[1];
  const float* attn_l = (const float*)d_in[2];
  const float* attn_r = (const float*)d_in[3];
  const int*   src    = (const int*)d_in[4];
  const int*   dst    = (const int*)d_in[5];
  float* out = (float*)d_out;

  char* ws = (char*)d_ws;
  __bf16* Xb = (__bf16*)ws;  ws += (size_t)N_NODES * IN_DIM * sizeof(__bf16);
  __bf16* Wb = (__bf16*)ws;  ws += (size_t)HD * IN_DIM * sizeof(__bf16);
  float*  ft = (float*)ws;   ws += (size_t)N_NODES * HD * sizeof(float);
  float*  a1 = (float*)ws;   ws += (size_t)N_NODES * HEADS * sizeof(float);
  float*  a2 = (float*)ws;   ws += (size_t)N_NODES * HEADS * sizeof(float);
  float* den = (float*)ws;

  // 1. bf16 copies of x and W
  cvt_to_bf16<<<2048, 256, 0, stream>>>(x,  Xb, N_NODES * IN_DIM);
  cvt_to_bf16<<<64,   256, 0, stream>>>(W,  Wb, HD * IN_DIM);

  // 2. ft = x @ W^T  (25000 waves, each a 16x64 strip; 8 waves/block)
  gat_gemm_wmma<<<3125, 256, 0, stream>>>(Xb, Wb, ft);

  // 3. per-node per-head scalars
  attn_scores<<<(N_NODES * HEADS + 255) / 256, 256, 0, stream>>>(ft, attn_l, attn_r, a1, a2);

  // 4. zero accumulators (d_out is poisoned by harness)
  zero_f32<<<2048, 256, 0, stream>>>(out, N_NODES * HD);
  zero_f32<<<256,  256, 0, stream>>>(den, N_NODES * HEADS);

  // 5. edge gather / atomic scatter (wave per edge)
  edge_scatter<<<N_EDGES / 8, 256, 0, stream>>>(ft, a1, a2, src, dst, out, den);

  // 6. normalize in place
  normalize_out<<<(N_NODES * HD + 255) / 256, 256, 0, stream>>>(out, den);
}
